// MPNNModel_59004260713106
// MI455X (gfx1250) — matrix operand order repaired and checked by reading, not compiled, not run on hardware
//
#include <hip/hip_runtime.h>

// ---------------------------------------------------------------------------
// MPNN (NNConv x3) inference for MI455X / gfx1250, wave32 + WMMA f16->f32.
//
// Refactor: per-edge dynamic-weight matmul == one dense GEMM per layer:
//   msg = Aaug[E,320] @ Wstack[320,64]
//   Aaug[e, d*64+h] = edge_attr[e,d] * h[src[e],h] (d<4);  Aaug[e,256+h] = h[src[e],h]
// Aaug is never materialized: each wave builds its A fragments in-register
// from gathered h_f16 rows (scale is constant per 32-wide k-slab: d = kk>>1).
// Scatter-add into agg is fused into the GEMM epilogue via f32 atomics;
// agg is pre-initialized by the root transform h @ root_W + conv_b.
// ---------------------------------------------------------------------------

typedef _Float16 half_t;
typedef __attribute__((ext_vector_type(16))) _Float16 v16h;
typedef __attribute__((ext_vector_type(8)))  _Float16 v8h;
typedef __attribute__((ext_vector_type(8)))  float    v8f;

#define N_NODES  8192
#define N_EDGES  16384
#define IN_CH    32
#define HID      64
#define OUT_CH   2
#define EDGE_DIM 4
#define N_LAYERS 3
#define N_GRAPHS 256
#define KAUG     320   // 5*HID : 4 edge-attr blocks + bias block
#define BN_EPS   1e-5f

// -------------------------- WMMA fragment loaders --------------------------
// A (16x32 f16, row-major, row stride = `stride` halves):
//   lanes 0-15 : row = lane,    halves 0-7 -> K=k0+0..7,  halves 8-15 -> K=k0+16..23
//   lanes 16-31: row = lane-16, halves 0-7 -> K=k0+8..15, halves 8-15 -> K=k0+24..31
__device__ __forceinline__ v16h load_a_frag(const half_t* A, int stride,
                                            int rowBase, int k0, int lane) {
    int r  = rowBase + (lane & 15);
    int kb = k0 + ((lane < 16) ? 0 : 8);
    const half_t* p = A + (size_t)r * stride + kb;
    v8h lo = *(const v8h*)(p);
    v8h hi = *(const v8h*)(p + 16);
    v16h a;
#pragma unroll
    for (int i = 0; i < 8; ++i) { a[i] = lo[i]; a[8 + i] = hi[i]; }
    return a;
}

// B (32x16 f16) from a K-contiguous (column-major) buffer BT[n*stride + k]:
//   lanes 0-15 : col = lane,    halves 0-7 -> K=k0+0..7,  halves 8-15 -> K=k0+16..23
//   lanes 16-31: col = lane-16, halves 0-7 -> K=k0+8..15, halves 8-15 -> K=k0+24..31
__device__ __forceinline__ v16h load_b_frag(const half_t* BT, int stride,
                                            int colBase, int k0, int lane) {
    int n  = colBase + (lane & 15);
    int kb = k0 + ((lane < 16) ? 0 : 8);
    const half_t* p = BT + (size_t)n * stride + kb;
    v8h lo = *(const v8h*)(p);
    v8h hi = *(const v8h*)(p + 16);
    v16h b;
#pragma unroll
    for (int i = 0; i < 8; ++i) { b[i] = lo[i]; b[8 + i] = hi[i]; }
    return b;
}

// ------------------------------ prep kernels -------------------------------
__global__ void k_cvt_f16(const float* __restrict__ s, half_t* __restrict__ d, int n) {
    int i = blockIdx.x * 256 + threadIdx.x;
    if (i < n) d[i] = (half_t)s[i];
}

// lin_in_W [32,64] -> linWT[n*32 + k]
__global__ void k_prep_linW(const float* __restrict__ W, half_t* __restrict__ WT) {
    int i = blockIdx.x * 256 + threadIdx.x;
    if (i < IN_CH * HID) {
        int n = i / IN_CH, k = i % IN_CH;
        WT[i] = (half_t)W[k * HID + n];
    }
}

// root_W [3,64,64] -> rootWT[l][n*64 + k]
__global__ void k_prep_rootW(const float* __restrict__ W, half_t* __restrict__ WT) {
    int i = blockIdx.x * 256 + threadIdx.x;
    if (i < N_LAYERS * HID * HID) {
        int l = i >> 12, r = i & 4095, n = r >> 6, k = r & 63;
        WT[i] = (half_t)W[(l << 12) + (k << 6) + n];
    }
}

// edge_W [3,4,4096], edge_b [3,4096] -> WstackT[l][o*320 + c]
//   c = d*64+h (d<4) -> edge_W[l,d,h*64+o] ;  c = 256+h -> edge_b[l,h*64+o]
__global__ void k_prep_Wstack(const float* __restrict__ eW, const float* __restrict__ eb,
                              half_t* __restrict__ WT) {
    int i = blockIdx.x * 256 + threadIdx.x;
    if (i < N_LAYERS * HID * KAUG) {
        int l = i / (HID * KAUG);
        int r = i % (HID * KAUG);
        int o = r / KAUG, c = r % KAUG;
        float v;
        if (c < 4 * HID) {
            int d = c >> 6, h = c & 63;
            v = eW[l * (EDGE_DIM * HID * HID) + d * (HID * HID) + h * HID + o];
        } else {
            int h = c & 63;
            v = eb[l * (HID * HID) + h * HID + o];
        }
        WT[i] = (half_t)v;
    }
}

// ------------------------------ GEMM kernels -------------------------------
// h0 = x @ lin_in_W + b   (K=32, 1 k-step; one wave per 16-row M-tile, all 4 N-tiles)
__global__ void k_lin_in(const half_t* __restrict__ xh, const half_t* __restrict__ WT,
                         const float* __restrict__ bias,
                         float* __restrict__ hf, half_t* __restrict__ hh) {
    int lane = threadIdx.x;
    int mT = blockIdx.x;
    v16h a = load_a_frag(xh, IN_CH, mT * 16, 0, lane);
    v8f acc[4] = {};
#pragma unroll
    for (int nT = 0; nT < 4; ++nT) {
        v16h b = load_b_frag(WT, IN_CH, nT * 16, 0, lane);
        acc[nT] = __builtin_amdgcn_wmma_f32_16x16x32_f16(false, a, false, b,
                                                         (short)0, acc[nT], false, false);
    }
    int mBase = mT * 16 + ((lane < 16) ? 0 : 8);
    int nLoc  = lane & 15;
#pragma unroll
    for (int nT = 0; nT < 4; ++nT) {
        int col = nT * 16 + nLoc;
        float bv = bias[col];
#pragma unroll
        for (int v = 0; v < 8; ++v) {
            float o = acc[nT][v] + bv;
            int idx = (mBase + v) * HID + col;
            hf[idx] = o;
            hh[idx] = (half_t)o;
        }
    }
}

// agg = h @ root_W[l] + conv_b[l]   (K=64, 2 k-steps; one wave per M-tile, 4 N-tiles)
__global__ void k_root(const half_t* __restrict__ hh, const half_t* __restrict__ rootWT,
                       const float* __restrict__ convb, float* __restrict__ agg, int l) {
    int lane = threadIdx.x;
    int mT = blockIdx.x;
    const half_t* WT = rootWT + l * HID * HID;
    v8f acc[4] = {};
#pragma unroll
    for (int kk = 0; kk < 2; ++kk) {
        v16h a = load_a_frag(hh, HID, mT * 16, kk * 32, lane);
#pragma unroll
        for (int nT = 0; nT < 4; ++nT) {
            v16h b = load_b_frag(WT, HID, nT * 16, kk * 32, lane);
            acc[nT] = __builtin_amdgcn_wmma_f32_16x16x32_f16(false, a, false, b,
                                                             (short)0, acc[nT], false, false);
        }
    }
    int mBase = mT * 16 + ((lane < 16) ? 0 : 8);
    int nLoc  = lane & 15;
#pragma unroll
    for (int nT = 0; nT < 4; ++nT) {
        int col = nT * 16 + nLoc;
        float bv = convb[l * HID + col];
#pragma unroll
        for (int v = 0; v < 8; ++v)
            agg[(mBase + v) * HID + col] = acc[nT][v] + bv;
    }
}

// msg GEMM with in-register Aaug build + fused scatter-add.
// One wave per 16-edge M-tile; K=320 (10 k-steps); all 4 N-tiles (40 WMMAs).
__global__ void k_msg(const half_t* __restrict__ hh, const int* __restrict__ srcIdx,
                      const float* __restrict__ ea, const half_t* __restrict__ WstackT,
                      const int* __restrict__ dstIdx, float* __restrict__ agg, int l) {
    int lane = threadIdx.x;
    int mT = blockIdx.x;
    const half_t* WT = WstackT + l * HID * KAUG;

    // A-layout row owned by this lane (edge index) and its gathered h row + scales
    int rA = mT * 16 + (lane & 15);
    const half_t* hrow = hh + (size_t)srcIdx[rA] * HID;
    half_t sc[5];
#pragma unroll
    for (int d = 0; d < 4; ++d) sc[d] = (half_t)ea[rA * EDGE_DIM + d];
    sc[4] = (half_t)1.0f;

    v8f acc[4] = {};
#pragma unroll
    for (int kk = 0; kk < 10; ++kk) {
        // Aaug fragment: two contiguous 8-half runs of the h row, scaled.
        // Within slab [kk*32, kk*32+32) the block index d = kk>>1 (bias for kk>=8).
        int off = (lane < 16) ? 0 : 8;
        int c1 = kk * 32 + off;        // run 1: c1..c1+7
        int c2 = c1 + 16;              // run 2: c2..c2+7
        half_t s = sc[kk >> 1];
        v8h x1 = *(const v8h*)(hrow + (c1 & 63));
        v8h x2 = *(const v8h*)(hrow + (c2 & 63));
        v16h a;
#pragma unroll
        for (int i = 0; i < 8; ++i) { a[i] = x1[i] * s; a[8 + i] = x2[i] * s; }
#pragma unroll
        for (int nT = 0; nT < 4; ++nT) {
            v16h b = load_b_frag(WT, KAUG, nT * 16, kk * 32, lane);
            acc[nT] = __builtin_amdgcn_wmma_f32_16x16x32_f16(false, a, false, b,
                                                             (short)0, acc[nT], false, false);
        }
    }

    int eBase = mT * 16 + ((lane < 16) ? 0 : 8);
    int nLoc  = lane & 15;
#pragma unroll
    for (int v = 0; v < 8; ++v) {
        int dn = dstIdx[eBase + v];
#pragma unroll
        for (int nT = 0; nT < 4; ++nT)
            atomicAdd(&agg[dn * HID + nT * 16 + nLoc], acc[nT][v]);
    }
}

// --------------------------- elementwise kernels ---------------------------
__global__ void k_bnrelu(const float* __restrict__ agg,
                         const float* __restrict__ gamma, const float* __restrict__ beta,
                         const float* __restrict__ mean,  const float* __restrict__ var,
                         float* __restrict__ hf, half_t* __restrict__ hh, int l) {
    int i = blockIdx.x * 256 + threadIdx.x;
    if (i < N_NODES * HID) {
        int ch = i & 63;
        float o = (agg[i] - mean[l * HID + ch]) *
                  rsqrtf(var[l * HID + ch] + BN_EPS) * gamma[l * HID + ch] +
                  beta[l * HID + ch];
        o = fmaxf(o, 0.0f);
        hf[i] = o;
        hh[i] = (half_t)o;
    }
}

__global__ void k_zero(float* __restrict__ p, int n) {
    int i = blockIdx.x * 256 + threadIdx.x;
    if (i < n) p[i] = 0.0f;
}

__global__ void k_pool(const float* __restrict__ hf, const int* __restrict__ batch,
                       float* __restrict__ pooled) {
    int i = blockIdx.x * 256 + threadIdx.x;
    if (i < N_NODES * HID) {
        int nn = i >> 6, ch = i & 63;
        atomicAdd(&pooled[batch[nn] * HID + ch], hf[i]);
    }
}

__global__ void k_clf(const float* __restrict__ pooled, const float* __restrict__ W,
                      const float* __restrict__ b, float* __restrict__ out) {
    int g = blockIdx.x * 256 + threadIdx.x;
    if (g < N_GRAPHS) {
        float a0 = b[0], a1 = b[1];
#pragma unroll
        for (int h = 0; h < HID; ++h) {
            float p = pooled[g * HID + h];
            a0 += p * W[h * OUT_CH + 0];
            a1 += p * W[h * OUT_CH + 1];
        }
        out[g * OUT_CH + 0] = a0;
        out[g * OUT_CH + 1] = a1;
    }
}

// ------------------------------- launcher ----------------------------------
extern "C" void kernel_launch(void* const* d_in, const int* in_sizes, int n_in,
                              void* d_out, int out_size, void* d_ws, size_t ws_size,
                              hipStream_t stream) {
    const float* x     = (const float*)d_in[0];
    const int*   eidx  = (const int*)d_in[1];
    const int*   src   = eidx;              // edge_index[0]
    const int*   dst   = eidx + N_EDGES;    // edge_index[1]
    const float* ea    = (const float*)d_in[2];
    const int*   batch = (const int*)d_in[3];
    const float* linW  = (const float*)d_in[4];
    const float* linb  = (const float*)d_in[5];
    const float* eW    = (const float*)d_in[6];
    const float* eb    = (const float*)d_in[7];
    const float* rW    = (const float*)d_in[8];
    const float* cb    = (const float*)d_in[9];
    const float* bng   = (const float*)d_in[10];
    const float* bnb   = (const float*)d_in[11];
    const float* bnm   = (const float*)d_in[12];
    const float* bnv   = (const float*)d_in[13];
    const float* clfW  = (const float*)d_in[14];
    const float* clfb  = (const float*)d_in[15];
    float* out = (float*)d_out;

    // Workspace partition (256B-aligned; total ~5.5 MB)
    char* ws = (char*)d_ws;
    size_t off = 0;
    auto alloc = [&](size_t bytes) -> void* {
        void* p = (void*)(ws + off);
        off += (bytes + 255) & ~(size_t)255;
        return p;
    };
    float*  h_f32   = (float*)alloc((size_t)N_NODES * HID * 4);
    float*  agg     = (float*)alloc((size_t)N_NODES * HID * 4);
    float*  pooled  = (float*)alloc((size_t)N_GRAPHS * HID * 4);
    half_t* h_f16   = (half_t*)alloc((size_t)N_NODES * HID * 2);
    half_t* x_f16   = (half_t*)alloc((size_t)N_NODES * IN_CH * 2);
    half_t* linWT   = (half_t*)alloc((size_t)HID * IN_CH * 2);
    half_t* rootWT  = (half_t*)alloc((size_t)N_LAYERS * HID * HID * 2);
    half_t* WstackT = (half_t*)alloc((size_t)N_LAYERS * HID * KAUG * 2);
    (void)ws_size; (void)in_sizes; (void)n_in; (void)out_size;

    // --- weight prep (f32 -> f16, K-contiguous layouts) ---
    k_cvt_f16    <<<(N_NODES * IN_CH + 255) / 256, 256, 0, stream>>>(x, x_f16, N_NODES * IN_CH);
    k_prep_linW  <<<(IN_CH * HID + 255) / 256, 256, 0, stream>>>(linW, linWT);
    k_prep_rootW <<<(N_LAYERS * HID * HID + 255) / 256, 256, 0, stream>>>(rW, rootWT);
    k_prep_Wstack<<<(N_LAYERS * HID * KAUG + 255) / 256, 256, 0, stream>>>(eW, eb, WstackT);

    // --- input projection ---
    k_lin_in<<<N_NODES / 16, 32, 0, stream>>>(x_f16, linWT, linb, h_f32, h_f16);

    // --- 3 NNConv layers ---
    for (int l = 0; l < N_LAYERS; ++l) {
        k_root<<<N_NODES / 16, 32, 0, stream>>>(h_f16, rootWT, cb, agg, l);
        k_msg <<<N_EDGES / 16, 32, 0, stream>>>(h_f16, src, ea, WstackT, dst, agg, l);
        k_bnrelu<<<(N_NODES * HID + 255) / 256, 256, 0, stream>>>(agg, bng, bnb, bnm, bnv,
                                                                  h_f32, h_f16, l);
    }

    // --- global_add_pool + classifier ---
    k_zero<<<(N_GRAPHS * HID + 255) / 256, 256, 0, stream>>>(pooled, N_GRAPHS * HID);
    k_pool<<<(N_NODES * HID + 255) / 256, 256, 0, stream>>>(h_f32, batch, pooled);
    k_clf <<<(N_GRAPHS + 255) / 256, 256, 0, stream>>>(pooled, clfW, clfb, out);
}
